// SpatialOnlyDivisiveNorm_56916906606727
// MI455X (gfx1250) — matrix operand order repaired and checked by reading, not compiled, not run on hardware
//
#include <hip/hip_runtime.h>

// ---------------------------------------------------------------------------
// SpatialOnlyDivisiveNorm for MI455X (gfx1250, wave32, WMMA + TDM)
//
//   out = x^2 / ((conv2d(x^2, exp(-r/lambda)) * alpha / lambda^2 + k)^beta + 1e-6)
//
// conv[4096 x 3136] = E[4096 x 3136] @ Kmat[3136 x 3136], E = bf16(x*x),
// Kmat[(y',x'),(y,x)] = exp(-sqrt(dy^2+dx^2)/lambda)*alpha/lambda^2,
// zero beyond |dy|,|dx| <= 28.  Kmat is SYMMETRIC, so the GEMM B-operand is
// read as Kmat[n][k] (K-contiguous) -> WMMA B fragments are contiguous LDS
// runs (ds_load_b128), no 16-bit element gathers.
// Tiles are staged LDS-double-buffered via the Tensor Data Mover
// (tensor_load_to_lds + s_wait_tensorcnt) so DMA overlaps WMMA.
// ---------------------------------------------------------------------------

typedef __attribute__((ext_vector_type(16))) __bf16 v16bf;
typedef __attribute__((ext_vector_type(8)))  __bf16 v8bf;
typedef __attribute__((ext_vector_type(4)))  __bf16 v4bf;
typedef __attribute__((ext_vector_type(8)))  float  v8f;
typedef unsigned int u32x4 __attribute__((ext_vector_type(4)));
typedef int          i32x4 __attribute__((ext_vector_type(4)));
typedef int          i32x8 __attribute__((ext_vector_type(8)));

#define HDIM 56
#define HW   3136                 // 56*56
#define BCN  4096                 // 32*128
#define KPAD 28                   // kernel reach (57x57 support)

#define BM 128
#define BN 64
#define BK 32
#define LDS_STRIDE 56             // padded row stride in bf16 (112 B = 7*16 B)

#if __has_builtin(__builtin_amdgcn_tensor_load_to_lds)
#define USE_TDM 1
#else
#define USE_TDM 0
#endif

// ---------------------------------------------------------------------------
// Kernel 1: synthesize the symmetric 3136x3136 kernel matrix in bf16.
// grid = (ceil(HW/256), HW): blockIdx.y = k row, x covers n.
// ---------------------------------------------------------------------------
__global__ void build_kmat_kernel(__bf16* __restrict__ Kmat,
                                  const float* __restrict__ lamb_p,
                                  const float* __restrict__ alpha_p) {
    int n  = blockIdx.x * blockDim.x + threadIdx.x;
    int kk = blockIdx.y;
    if (n >= HW) return;

    int yp = kk / HDIM, xp = kk % HDIM;
    int y  = n  / HDIM, x  = n  % HDIM;
    int dy = y - yp, dx = x - xp;

    float v = 0.0f;
    if (dy >= -KPAD && dy <= KPAD && dx >= -KPAD && dx <= KPAD) {
        float lamb  = lamb_p[0];
        float alpha = alpha_p[0];
        float r     = sqrtf((float)(dy * dy + dx * dx));
        float denom = lamb + 1e-6f;
        v = __expf(-r / lamb) * alpha / (denom * denom);
    }
    Kmat[(size_t)kk * HW + n] = (__bf16)v;
}

// ---------------------------------------------------------------------------
// Kernel 2: E = bf16(x*x), vectorized 4-wide.
// ---------------------------------------------------------------------------
__global__ void square_to_bf16_kernel(const float4* __restrict__ x4,
                                      v4bf* __restrict__ E4, int n4) {
    int i = blockIdx.x * blockDim.x + threadIdx.x;
    if (i >= n4) return;
    float4 v = x4[i];
    v4bf e;
    e[0] = (__bf16)(v.x * v.x);
    e[1] = (__bf16)(v.y * v.y);
    e[2] = (__bf16)(v.z * v.z);
    e[3] = (__bf16)(v.w * v.w);
    E4[i] = e;
}

// ---------------------------------------------------------------------------
// TDM helper: 2D tile load Global -> LDS.
//   tile = tile_rows rows of 32 bf16 (64 B), row stride = HW elements,
//   LDS rows padded 64 B data + 48 B pad (pad_interval=16 DW, pad_amount=12 DW)
//   -> 112 B LDS stride == LDS_STRIDE bf16.
// ---------------------------------------------------------------------------
#if USE_TDM
__device__ __forceinline__ unsigned lds_byte_offset(const void* p) {
    return (unsigned)(unsigned long long)
        (const __attribute__((address_space(3))) char*)p;
}

__device__ __forceinline__ void tdm_load_tile(unsigned lds_addr,
                                              const __bf16* gptr,
                                              int tile_rows) {
    unsigned long long ga = (unsigned long long)gptr;
    u32x4 g0;
    g0[0] = 1u;                                        // count=1, no gather
    g0[1] = lds_addr;                                  // LDS byte address
    g0[2] = (unsigned)(ga & 0xFFFFFFFFu);              // global_addr[31:0]
    g0[3] = (unsigned)((ga >> 32) & 0x1FFFFFFu)        // global_addr[56:32]
          | (2u << 30);                                // type = 2 ("image")

    i32x8 g1;
    g1[0] = (1 << 16)      // data_size = 2 bytes
          | (1 << 20)      // pad_enable
          | (3 << 22)      // pad_interval: 16 DWORDs (= one 64 B row)
          | (11 << 25);    // pad_amount: 12 DWORDs (48 B) -> 112 B stride
    g1[1] = (HW & 0xFFFF) << 16;                 // tensor_dim0[15:0]
    g1[2] = (HW >> 16) & 0xFFFF;                 // tensor_dim0[31:16], dim1 lo=0
    g1[3] = 0x10 | (BK << 16);                   // tensor_dim1 hi (=1M), tile_dim0=32
    g1[4] = tile_rows & 0xFFFF;                  // tile_dim1, tile_dim2 = 0
    g1[5] = HW;                                  // tensor_dim0_stride[31:0]
    g1[6] = 0;
    g1[7] = 0;

    i32x4 z4 = {};
#if defined(__clang_major__) && (__clang_major__ >= 23)
    i32x8 z8 = {};
    __builtin_amdgcn_tensor_load_to_lds(g0, g1, z4, z4, z8, 0);
#else
    __builtin_amdgcn_tensor_load_to_lds(g0, g1, z4, z4, 0);
#endif
}
#endif // USE_TDM

// ---------------------------------------------------------------------------
// Kernel 3: double-buffered WMMA GEMM + fused divisive-norm epilogue.
//   grid = (HW/BN = 49, BCN/BM = 32), block = 256 threads = 8 waves.
//   Wave (wm, wn) owns a 32x32 patch = 2x2 v_wmma_f32_16x16x32_bf16 tiles.
// ---------------------------------------------------------------------------
__global__ __launch_bounds__(256)
void dnorm_gemm_kernel(const __bf16* __restrict__ E,
                       const __bf16* __restrict__ Kmat,
                       const float*  __restrict__ x,
                       float*        __restrict__ out,
                       const float*  __restrict__ beta_p,
                       const float*  __restrict__ kconst_p) {
    __shared__ __bf16 As[2][BM][LDS_STRIDE];   // 2 x 128 x 112B = 28 KB
    __shared__ __bf16 Bs[2][BN][LDS_STRIDE];   // 2 x  64 x 112B = 14 KB

    const int tid   = threadIdx.x;
    const int lane  = tid & 31;
    const int wave  = tid >> 5;        // 0..7
    const int wm    = wave >> 1;       // 0..3 : 32-row band within BM
    const int wn    = wave & 1;        // 0..1 : 32-col band within BN

    const int rowbase = blockIdx.y * BM;   // position in 4096 (B*C) rows
    const int nbase   = blockIdx.x * BN;   // position in 3136 spatial cols

    const int lrow  = lane & 15;       // M/N index within a 16-wide tile
    const int khalf = lane >> 4;       // K-half held by this lane

    v8f acc[2][2] = {};

    const int NT = HW / BK;            // 98 K-steps

#if USE_TDM
    // -------- stage tile 0 via the Tensor Data Mover (wave 0 only) --------
    if (wave == 0) {
        tdm_load_tile(lds_byte_offset(&As[0][0][0]),
                      E + (size_t)rowbase * HW, BM);
        tdm_load_tile(lds_byte_offset(&Bs[0][0][0]),
                      Kmat + (size_t)nbase * HW, BN);
    }
#else
    // -------- fallback: register-staged copy of tile 0 --------
    {
        int r0 = tid >> 2, c8 = (tid & 3) * 8;
        int r1 = (tid + 256) >> 2;
        uint4 a0 = *(const uint4*)(E + (size_t)(rowbase + r0) * HW + c8);
        uint4 a1 = *(const uint4*)(E + (size_t)(rowbase + r1) * HW + c8);
        uint4 b0 = *(const uint4*)(Kmat + (size_t)(nbase + (tid >> 2)) * HW + c8);
        *(uint4*)&As[0][r0][c8] = a0;
        *(uint4*)&As[0][r1][c8] = a1;
        *(uint4*)&Bs[0][tid >> 2][c8] = b0;
    }
#endif

    for (int t = 0; t < NT; ++t) {
        const int buf = t & 1;
        const int k0n = (t + 1) * BK;

        // -------- stage next tile into the other buffer (overlaps WMMA) ----
        if (t + 1 < NT) {
#if USE_TDM
            if (wave == 0) {
                tdm_load_tile(lds_byte_offset(&As[buf ^ 1][0][0]),
                              E + (size_t)rowbase * HW + k0n, BM);
                tdm_load_tile(lds_byte_offset(&Bs[buf ^ 1][0][0]),
                              Kmat + (size_t)nbase * HW + k0n, BN);
            }
#else
            int r0 = tid >> 2, c8 = (tid & 3) * 8;
            int r1 = (tid + 256) >> 2;
            uint4 a0 = *(const uint4*)(E + (size_t)(rowbase + r0) * HW + k0n + c8);
            uint4 a1 = *(const uint4*)(E + (size_t)(rowbase + r1) * HW + k0n + c8);
            uint4 b0 = *(const uint4*)(Kmat + (size_t)(nbase + (tid >> 2)) * HW + k0n + c8);
            *(uint4*)&As[buf ^ 1][r0][c8] = a0;
            *(uint4*)&As[buf ^ 1][r1][c8] = a1;
            *(uint4*)&Bs[buf ^ 1][tid >> 2][c8] = b0;
#endif
        }

#if USE_TDM
        // current tile's 2 TDM ops done; next tile's 2 may remain in flight
        if (t + 1 < NT) __builtin_amdgcn_s_wait_tensorcnt(2);
        else            __builtin_amdgcn_s_wait_tensorcnt(0);
#endif
        __syncthreads();

        // -------- gather fragments (all 16 B-aligned contiguous runs) ------
        // A (16x32): lane = row M; v0..3 = K[kh*8..+7], v4..7 = K[16+kh*8..+7]
        v16bf a[2], b[2];
        #pragma unroll
        for (int mt = 0; mt < 2; ++mt) {
            int row = wm * 32 + mt * 16 + lrow;
            v8bf lo = *(const v8bf*)&As[buf][row][khalf * 8];
            v8bf hi = *(const v8bf*)&As[buf][row][16 + khalf * 8];
            a[mt] = __builtin_shufflevector(lo, hi, 0, 1, 2, 3, 4, 5, 6, 7,
                                            8, 9, 10, 11, 12, 13, 14, 15);
        }
        // B (32x16): lane = col N; v0..3 = K[kh*8..+7], v4..7 = K[16+kh*8..+7]
        #pragma unroll
        for (int nt = 0; nt < 2; ++nt) {
            int col = wn * 32 + nt * 16 + lrow;
            v8bf lo = *(const v8bf*)&Bs[buf][col][khalf * 8];
            v8bf hi = *(const v8bf*)&Bs[buf][col][16 + khalf * 8];
            b[nt] = __builtin_shufflevector(lo, hi, 0, 1, 2, 3, 4, 5, 6, 7,
                                            8, 9, 10, 11, 12, 13, 14, 15);
        }

        // -------- 2x2 WMMA rank-32 updates --------
        #pragma unroll
        for (int mt = 0; mt < 2; ++mt)
            #pragma unroll
            for (int nt = 0; nt < 2; ++nt)
                acc[mt][nt] = __builtin_amdgcn_wmma_f32_16x16x32_bf16(
                    false, a[mt], false, b[nt],
                    (short)0, acc[mt][nt], false, false);

        __syncthreads();   // everyone done with buf before it is re-staged
    }

    // -------- fused epilogue: out = x^2 / ((conv + k)^beta + 1e-6) --------
    // C/D layout: lane 0-15 -> N=lane, M=r ; lane 16-31 -> N=lane-16, M=r+8
    const float beta = beta_p[0];
    const float kc   = kconst_p[0];
    #pragma unroll
    for (int mt = 0; mt < 2; ++mt) {
        #pragma unroll
        for (int nt = 0; nt < 2; ++nt) {
            int gn = nbase + wn * 32 + nt * 16 + lrow;
            #pragma unroll
            for (int r = 0; r < 8; ++r) {
                int gm = rowbase + wm * 32 + mt * 16 + khalf * 8 + r;
                size_t idx = (size_t)gm * HW + gn;
                float xv   = x[idx];
                float conv = acc[mt][nt][r];
                float divisor = __powf(conv + kc, beta);
                out[idx] = xv * xv / (divisor + 1e-6f);
            }
        }
    }
}

// ---------------------------------------------------------------------------
// Host-side launch (graph-capture safe: stream-only, no allocation).
// Inputs (setup_inputs order): x, xy_lamb, alpha, beta, k.
// Workspace: [ Kmat bf16 : 19.7 MB ][ E bf16 : 25.7 MB ]  (~46 MB total)
// ---------------------------------------------------------------------------
extern "C" void kernel_launch(void* const* d_in, const int* in_sizes, int n_in,
                              void* d_out, int out_size, void* d_ws, size_t ws_size,
                              hipStream_t stream) {
    const float* x     = (const float*)d_in[0];
    const float* lamb  = (const float*)d_in[1];
    const float* alpha = (const float*)d_in[2];
    const float* beta  = (const float*)d_in[3];
    const float* kcst  = (const float*)d_in[4];
    float* out = (float*)d_out;

    __bf16* Kmat = (__bf16*)d_ws;
    size_t kmat_bytes = ((size_t)HW * HW * sizeof(__bf16) + 255) & ~(size_t)255;
    __bf16* E = (__bf16*)((char*)d_ws + kmat_bytes);

    // 1) build the symmetric kernel matrix (9.83M elements)
    dim3 kgrid((HW + 255) / 256, HW);
    build_kmat_kernel<<<kgrid, 256, 0, stream>>>(Kmat, lamb, alpha);

    // 2) E = bf16(x*x)
    int n4 = (BCN * HW) / 4;
    square_to_bf16_kernel<<<(n4 + 255) / 256, 256, 0, stream>>>(
        (const float4*)x, (v4bf*)E, n4);

    // 3) double-buffered WMMA GEMM + epilogue
    dim3 grid(HW / BN, BCN / BM);   // (49, 32)
    dnorm_gemm_kernel<<<grid, 256, 0, stream>>>(E, Kmat, x, out, beta, kcst);
}
